// ATSBlock_46084999086121
// MI455X (gfx1250) — matrix operand order repaired and checked by reading, not compiled, not run on hardware
//
#include <hip/hip_runtime.h>
#include <math.h>

// ---------------------------------------------------------------------------
// ATS transformer block for MI455X (gfx1250), wave32 + WMMA bf16/f32.
// ---------------------------------------------------------------------------

typedef unsigned short ushort_t;
typedef __attribute__((ext_vector_type(16))) __bf16       v16bf;
typedef __attribute__((ext_vector_type(8)))  float        v8f;
typedef __attribute__((ext_vector_type(4)))  unsigned int u32x4;

#define B_        16
#define N_        577
#define C_        768
#define H_        12
#define D_        64
#define NSEL      197      // output tokens per batch (n_ref_tokens)
#define NKEEP     196      // NSEL - 1
#define NTOK1     576      // N_ - 1
#define NPAD      608      // token dim padded to multiple of 32 (19*32)
#define ROWS_ALL  (B_ * N_)    // 9232
#define ROWS_SEL  (B_ * NSEL)  // 3152
#define SCALE_QK  0.125f       // D^-0.5
#define EPS_SM    1e-6f
#define LN_EPS    1e-5f

// ---------------- bf16 helpers (memory format = ushort) --------------------
__device__ __forceinline__ ushort_t f2bf(float f) {
    unsigned u = __builtin_bit_cast(unsigned, f);
    unsigned r = u + 0x7FFFu + ((u >> 16) & 1u);   // round to nearest even
    return (ushort_t)(r >> 16);
}
__device__ __forceinline__ float bf2f(ushort_t h) {
    unsigned u = ((unsigned)h) << 16;
    return __builtin_bit_cast(float, u);
}

union Frag {
    v16bf v;
    u32x4 q[2];
};
static_assert(sizeof(Frag) == 32, "frag size");

__device__ __forceinline__ v8f wmma_bf16(v16bf a, v16bf b, v8f c) {
    return __builtin_amdgcn_wmma_f32_16x16x32_bf16(
        /*neg_a=*/false, a, /*neg_b=*/false, b,
        /*c_mod=*/(short)0, c, /*reuse_a=*/false, /*reuse_b=*/false);
}

// A fragment: 16x32 bf16, lane = row (l&15), half=l>>4.
// VGPR v : K = half*16 + 2v,2v+1 (same order as B)
// => one contiguous 32-byte run at k0+half*16.
__device__ __forceinline__ v16bf load_a_frag(const ushort_t* base, int lda,
                                             int row0, int k0, int max_row) {
    int lane = threadIdx.x & 31, m = lane & 15, half = lane >> 4;
    Frag f;
    if (row0 + 16 <= max_row) {          // wave-uniform: whole tile valid
        const ushort_t* p = base + (long)(row0 + m) * lda + k0;
        f.q[0] = *(const u32x4*)(p + half * 16);
        f.q[1] = *(const u32x4*)(p + half * 16 + 8);
    } else {
        int row = row0 + m;
        if (row < max_row) {
            const ushort_t* p = base + (long)row * lda + k0;
            f.q[0] = *(const u32x4*)(p + half * 16);
            f.q[1] = *(const u32x4*)(p + half * 16 + 8);
        } else {
            u32x4 z = {0, 0, 0, 0};
            f.q[0] = z; f.q[1] = z;
        }
    }
    return f.v;
}

// B fragment: 32x16 bf16, lane = col (l&15), half=l>>4.
// VGPR v : K = half*16 + 2v,2v+1  => one contiguous 32-byte run at k0+half*16.
// BT is stored K-contiguous per output column (i.e. transposed weights).
__device__ __forceinline__ v16bf load_b_frag(const ushort_t* baseT, int ldb,
                                             int col0, int k0, int max_col) {
    int lane = threadIdx.x & 31, n = lane & 15, half = lane >> 4;
    Frag f;
    if (col0 + 16 <= max_col) {          // wave-uniform: whole tile valid
        const ushort_t* p = baseT + (long)(col0 + n) * ldb + k0 + half * 16;
        f.q[0] = *(const u32x4*)(p);
        f.q[1] = *(const u32x4*)(p + 8);
    } else {
        int col = col0 + n;
        if (col < max_col) {
            const ushort_t* p = baseT + (long)col * ldb + k0 + half * 16;
            f.q[0] = *(const u32x4*)(p);
            f.q[1] = *(const u32x4*)(p + 8);
        } else {
            u32x4 z = {0, 0, 0, 0};
            f.q[0] = z; f.q[1] = z;
        }
    }
    return f.v;
}

#define LOADA(dst, p, k0)                                  \
    do {                                                   \
        (dst).q[0] = *(const u32x4*)((p) + (k0));          \
        (dst).q[1] = *(const u32x4*)((p) + (k0) + 8);      \
    } while (0)
#define LOADB(dst, p, k0)                                  \
    do {                                                   \
        (dst).q[0] = *(const u32x4*)((p) + (k0));          \
        (dst).q[1] = *(const u32x4*)((p) + (k0) + 8);      \
    } while (0)

// -------------------- generic 8-wave (256 thr) GEMM core -------------------
// Block tile 128(M) x 64(N): waves 4x2, each wave 2x2 tiles of 16x16.
struct Acc {
    v8f c[2][2];
};

#define LOAD_SET(s, k0)                                    \
    do {                                                   \
        LOADA(a0##s, pa0, (k0)); LOADA(a1##s, pa1, (k0));  \
        LOADB(b0##s, pb0, (k0)); LOADB(b1##s, pb1, (k0));  \
    } while (0)
#define WMMA_SET(s)                                                    \
    do {                                                               \
        acc.c[0][0] = wmma_bf16(a0##s.v, b0##s.v, acc.c[0][0]);        \
        acc.c[0][1] = wmma_bf16(a0##s.v, b1##s.v, acc.c[0][1]);        \
        acc.c[1][0] = wmma_bf16(a1##s.v, b0##s.v, acc.c[1][0]);        \
        acc.c[1][1] = wmma_bf16(a1##s.v, b1##s.v, acc.c[1][1]);        \
    } while (0)

__device__ __forceinline__ void gemm_core(const ushort_t* A, int lda, int arow0, int amax,
                                          const ushort_t* BT, int ldb, int bcol0, int bmax,
                                          int Kdim, Acc& acc) {
    v8f z = {0, 0, 0, 0, 0, 0, 0, 0};
    acc.c[0][0] = z; acc.c[0][1] = z; acc.c[1][0] = z; acc.c[1][1] = z;
    int lane = threadIdx.x & 31, lm = lane & 15, half = lane >> 4;

    if (arow0 + 32 <= amax && bcol0 + 32 <= bmax) {
        // ---- fast path: whole wave tile in bounds; unguarded, copy-free
        //      ping-pong pipeline (two K-steps per iteration, no rotation movs)
        const ushort_t* pa0 = A + (long)(arow0 + lm) * lda + half * 16;
        const ushort_t* pa1 = pa0 + (long)16 * lda;
        const ushort_t* pb0 = BT + (long)(bcol0 + lm) * ldb + half * 16;
        const ushort_t* pb1 = pb0 + (long)16 * ldb;
        Frag a0A, a1A, b0A, b1A;   // set A
        Frag a0B, a1B, b0B, b1B;   // set B
        LOAD_SET(A, 0);
        int k0 = 32;
        for (; k0 + 32 < Kdim; k0 += 64) {
            LOAD_SET(B, k0);
            __builtin_prefetch(pb0 + k0 + 64, 0, 1);   // speculative: next K-slabs
            __builtin_prefetch(pa0 + k0 + 64, 0, 1);
            WMMA_SET(A);
            LOAD_SET(A, k0 + 32);
            WMMA_SET(B);
        }
        if (k0 < Kdim) {           // one trailing K-step (even step count)
            LOAD_SET(B, k0);
            WMMA_SET(A);
            WMMA_SET(B);
        } else {
            WMMA_SET(A);
        }
    } else {
        // ---- slow path: edge block, tile-granular guards
        for (int k0 = 0; k0 < Kdim; k0 += 32) {
            v16bf a0 = load_a_frag(A, lda, arow0, k0, amax);
            v16bf a1 = load_a_frag(A, lda, arow0 + 16, k0, amax);
            v16bf b0 = load_b_frag(BT, ldb, bcol0, k0, bmax);
            v16bf b1 = load_b_frag(BT, ldb, bcol0 + 16, k0, bmax);
            acc.c[0][0] = wmma_bf16(a0, b0, acc.c[0][0]);
            acc.c[0][1] = wmma_bf16(a0, b1, acc.c[0][1]);
            acc.c[1][0] = wmma_bf16(a1, b0, acc.c[1][0]);
            acc.c[1][1] = wmma_bf16(a1, b1, acc.c[1][1]);
        }
    }
}

// ---------------------------- utility kernels ------------------------------
__global__ void k_cvt_T(const float* __restrict__ W, ushort_t* __restrict__ WT,
                        int K, int N) {
    long i = (long)blockIdx.x * blockDim.x + threadIdx.x;
    long tot = (long)K * N;
    if (i >= tot) return;
    int k = (int)(i / N), n = (int)(i % N);
    WT[(long)n * K + k] = f2bf(W[i]);
}

__global__ void k_zero_us(ushort_t* p, long n) {
    long i = (long)blockIdx.x * blockDim.x + threadIdx.x;
    long stride = (long)gridDim.x * blockDim.x;
    for (; i < n; i += stride) p[i] = 0;
}

// layer norm: one block per row, fp32 in -> bf16 out
__global__ __launch_bounds__(256) void k_layernorm(const float* __restrict__ X,
                                                   const float* __restrict__ g,
                                                   const float* __restrict__ beta,
                                                   ushort_t* __restrict__ out, int C) {
    int row = blockIdx.x;
    const float* x = X + (long)row * C;
    float s = 0.f, s2 = 0.f;
    for (int c = threadIdx.x; c < C; c += 256) {
        float v = x[c];
        s += v; s2 += v * v;
    }
    for (int m = 16; m; m >>= 1) {
        s += __shfl_xor(s, m, 32);
        s2 += __shfl_xor(s2, m, 32);
    }
    __shared__ float r1[8], r2[8];
    int wid = threadIdx.x >> 5;
    if ((threadIdx.x & 31) == 0) { r1[wid] = s; r2[wid] = s2; }
    __syncthreads();
    if (threadIdx.x == 0) {
        float a = 0.f, b = 0.f;
        for (int w = 0; w < 8; ++w) { a += r1[w]; b += r2[w]; }
        r1[0] = a; r2[0] = b;
    }
    __syncthreads();
    float mu = r1[0] / C;
    float var = r2[0] / C - mu * mu;
    float inv = rsqrtf(var + LN_EPS);
    for (int c = threadIdx.x; c < C; c += 256)
        out[(long)row * C + c] = f2bf((x[c] - mu) * inv * g[c] + beta[c]);
}

// ------------------------------ QKV GEMM -----------------------------------
// xn[9232,768] @ wqkvT -> scatter q/k bf16 [b,h,n,d], vT bf16 [b,h,d,nPAD], *policy
__global__ __launch_bounds__(256) void k_gemm_qkv(const ushort_t* __restrict__ xn,
                                                  const ushort_t* __restrict__ wT,
                                                  const float* __restrict__ policy,
                                                  ushort_t* __restrict__ qb,
                                                  ushort_t* __restrict__ kb,
                                                  ushort_t* __restrict__ vT) {
    int wid = threadIdx.x >> 5, lane = threadIdx.x & 31, lm = lane & 15, half = lane >> 4;
    int wm = wid & 3, wn = wid >> 2;
    int arow0 = blockIdx.x * 128 + wm * 32;
    int bcol0 = blockIdx.y * 64 + wn * 32;
    Acc acc;
    gemm_core(xn, C_, arow0, ROWS_ALL, wT, C_, bcol0, 3 * C_, C_, acc);
    for (int ti = 0; ti < 2; ++ti)
        for (int tj = 0; tj < 2; ++tj)
            for (int r = 0; r < 8; ++r) {
                int row = arow0 + ti * 16 + r + 8 * half;
                int col = bcol0 + tj * 16 + lm;
                if (row >= ROWS_ALL) continue;
                int bb = row / N_, n = row % N_;
                float val = acc.c[ti][tj][r] * policy[row];
                int which = col / C_, cc = col % C_;
                int hh = cc >> 6, dd = cc & 63;
                long base = (long)(bb * H_ + hh);
                if (which == 0)      qb[(base * N_ + n) * D_ + dd] = f2bf(val);
                else if (which == 1) kb[(base * N_ + n) * D_ + dd] = f2bf(val);
                else                 vT[(base * D_ + dd) * NPAD + n] = f2bf(val);
            }
}

// --------------- fused attention scores + softmax_with_policy --------------
// One block per (b, h, row-tile of 16). Each of 8 waves covers col tiles
// ct = wid, wid+8, ... (38 tiles of 16 => NPAD=608 columns).
__global__ __launch_bounds__(256) void k_scores_softmax(const ushort_t* __restrict__ qb,
                                                        const ushort_t* __restrict__ kb,
                                                        const float* __restrict__ policy,
                                                        ushort_t* __restrict__ attn,
                                                        float* __restrict__ attn_cls) {
    int mtile = blockIdx.x, h = blockIdx.y, b = blockIdx.z;
    const ushort_t* Q = qb + (long)(b * H_ + h) * N_ * D_;
    const ushort_t* Kt = kb + (long)(b * H_ + h) * N_ * D_;
    int wid = threadIdx.x >> 5, lane = threadIdx.x & 31, lm = lane & 15, half = lane >> 4;
    int row0 = mtile * 16;
    const int NTC = NPAD / 16;  // 38
    v8f acc[5];
    int cts[5];
    int cnt = 0;
    v8f z = {0, 0, 0, 0, 0, 0, 0, 0};
    for (int ct = wid; ct < NTC; ct += 8) { cts[cnt] = ct; acc[cnt] = z; ++cnt; }

    for (int k0 = 0; k0 < D_; k0 += 32) {
        v16bf a = load_a_frag(Q, D_, row0, k0, N_);
        for (int i = 0; i < cnt; ++i) {
            v16bf bf = load_b_frag(Kt, D_, cts[i] * 16, k0, N_);  // B = K^T (col=token)
            acc[i] = wmma_bf16(a, bf, acc[i]);
        }
    }

    __shared__ float red[16][8];
    float rmax[8];
    for (int r = 0; r < 8; ++r) {
        float m = -1e30f;
        for (int i = 0; i < cnt; ++i) {
            int col = cts[i] * 16 + lm;
            if (col < N_) m = fmaxf(m, acc[i][r] * SCALE_QK);
        }
        for (int s = 8; s; s >>= 1) m = fmaxf(m, __shfl_xor(m, s, 32));
        rmax[r] = m;
    }
    if (lm == 0)
        for (int r = 0; r < 8; ++r) red[r + 8 * half][wid] = rmax[r];
    __syncthreads();
    for (int r = 0; r < 8; ++r) {
        float m = -1e30f;
        for (int w = 0; w < 8; ++w) m = fmaxf(m, red[r + 8 * half][w]);
        rmax[r] = m;
    }
    __syncthreads();

    float rsum[8];
    for (int r = 0; r < 8; ++r) {
        int grow = row0 + r + 8 * half;
        float s = 0.f;
        for (int i = 0; i < cnt; ++i) {
            int col = cts[i] * 16 + lm;
            float e = 0.f;
            if (col < N_) {
                float p = policy[b * N_ + col];
                float ap = p + (1.f - p) * ((col == grow) ? 1.f : 0.f);
                e = expf(acc[i][r] * SCALE_QK - rmax[r]) * ap;
            }
            acc[i][r] = e;
            s += e;
        }
        for (int sm = 8; sm; sm >>= 1) s += __shfl_xor(s, sm, 32);
        rsum[r] = s;
    }
    if (lm == 0)
        for (int r = 0; r < 8; ++r) red[r + 8 * half][wid] = rsum[r];
    __syncthreads();
    for (int r = 0; r < 8; ++r) {
        float s = 0.f;
        for (int w = 0; w < 8; ++w) s += red[r + 8 * half][w];
        rsum[r] = s;
    }

    const float epsn = EPS_SM / (float)N_;
    for (int r = 0; r < 8; ++r) {
        int grow = row0 + r + 8 * half;
        if (grow >= N_) continue;
        float denom = rsum[r] + EPS_SM;
        for (int i = 0; i < cnt; ++i) {
            int col = cts[i] * 16 + lm;
            float val = (col < N_) ? (acc[i][r] + epsn) / denom : 0.f;  // zero pad cols
            attn[((long)(b * H_ + h) * N_ + grow) * NPAD + col] = f2bf(val);
            if (grow == 0) attn_cls[(long)(b * H_ + h) * NPAD + col] = val;
        }
    }
}

// ------------------------------ attn @ v -----------------------------------
__global__ __launch_bounds__(256) void k_attn_v(const ushort_t* __restrict__ attn,
                                                const ushort_t* __restrict__ vT,
                                                ushort_t* __restrict__ Ofull) {
    int h = blockIdx.y, b = blockIdx.z;
    const ushort_t* A = attn + (long)(b * H_ + h) * N_ * NPAD;
    const ushort_t* Bm = vT + (long)(b * H_ + h) * D_ * NPAD;
    int wid = threadIdx.x >> 5, lane = threadIdx.x & 31, lm = lane & 15, half = lane >> 4;
    int wm = wid & 3, wn = wid >> 2;
    int arow0 = blockIdx.x * 128 + wm * 32;
    int bcol0 = wn * 32;  // N = 64 exactly
    Acc acc;
    gemm_core(A, NPAD, arow0, N_, Bm, NPAD, bcol0, D_, NPAD, acc);
    for (int ti = 0; ti < 2; ++ti)
        for (int tj = 0; tj < 2; ++tj)
            for (int r = 0; r < 8; ++r) {
                int tok = arow0 + ti * 16 + r + 8 * half;
                int col = bcol0 + tj * 16 + lm;
                if (tok < N_)
                    Ofull[((long)b * N_ + tok) * C_ + h * D_ + col] = f2bf(acc.c[ti][tj][r]);
            }
}

// ------------------------- token selection (per batch) ---------------------
__device__ __forceinline__ float blk_red_sum(float v, float* sh8) {
    for (int m = 16; m; m >>= 1) v += __shfl_xor(v, m, 32);
    int wid = threadIdx.x >> 5;
    if ((threadIdx.x & 31) == 0) sh8[wid] = v;
    __syncthreads();
    if (threadIdx.x == 0) {
        float t = 0.f;
        for (int w = 0; w < 8; ++w) t += sh8[w];
        sh8[0] = t;
    }
    __syncthreads();
    float t = sh8[0];
    __syncthreads();
    return t;
}
__device__ __forceinline__ float blk_red_min(float v, float* sh8) {
    for (int m = 16; m; m >>= 1) v = fminf(v, __shfl_xor(v, m, 32));
    int wid = threadIdx.x >> 5;
    if ((threadIdx.x & 31) == 0) sh8[wid] = v;
    __syncthreads();
    if (threadIdx.x == 0) {
        float t = 1e30f;
        for (int w = 0; w < 8; ++w) t = fminf(t, sh8[w]);
        sh8[0] = t;
    }
    __syncthreads();
    float t = sh8[0];
    __syncthreads();
    return t;
}

__global__ __launch_bounds__(256) void k_select(const float* __restrict__ attn_cls,
                                                const ushort_t* __restrict__ vT,
                                                int* __restrict__ row_map,
                                                float* __restrict__ pol_ws,
                                                float* __restrict__ out_np) {
    int b = blockIdx.x, tid = threadIdx.x;
    __shared__ float s_sig[NTOK1];
    __shared__ float s_sorted[NTOK1];
    __shared__ int s_idx[NTOK1];
    __shared__ float s_cdf[NTOK1];
    __shared__ float s_r8[8];
    __shared__ float s_ys[NKEEP];
    __shared__ int s_pick[NKEEP];
    __shared__ int s_u[NKEEP];
    __shared__ int s_u2[NKEEP];

    // sig = (sum_h attn[b,h,0,tok]) * ||v[b,:,tok,:]||  for tok = 1..576
    for (int n = tid; n < NTOK1; n += 256) {
        int tok = n + 1;
        float cls = 0.f, vv = 0.f;
        for (int h = 0; h < H_; ++h) {
            cls += attn_cls[(long)(b * H_ + h) * NPAD + tok];
            const ushort_t* vp = vT + ((long)(b * H_ + h) * D_) * NPAD + tok;
            for (int d = 0; d < D_; ++d) {
                float v = bf2f(vp[(long)d * NPAD]);
                vv += v * v;
            }
        }
        s_sig[n] = cls * sqrtf(vv);
    }
    __syncthreads();
    float part = 0.f;
    for (int n = tid; n < NTOK1; n += 256) part += s_sig[n];
    float total = blk_red_sum(part, s_r8);
    for (int n = tid; n < NTOK1; n += 256) s_sig[n] /= total;
    __syncthreads();

    // stable ascending rank sort (matches jnp.sort/argsort)
    for (int i = tid; i < NTOK1; i += 256) {
        float si = s_sig[i];
        int r = 0;
        for (int j = 0; j < NTOK1; ++j) {
            float sj = s_sig[j];
            r += (sj < si) || (sj == si && j < i);
        }
        s_sorted[r] = si;
        s_idx[r] = i;
    }
    __syncthreads();
    if (tid == 0) {
        float c = 0.f;
        for (int j = 0; j < NTOK1; ++j) { c += s_sorted[j]; s_cdf[j] = c; }
    }
    __syncthreads();
    float cmin = s_cdf[0], cmax = s_cdf[NTOK1 - 1];
    for (int j = tid; j < NTOK1; j += 256) s_cdf[j] = (s_cdf[j] - cmin) / (cmax - cmin);
    __syncthreads();

    float mn = 1e30f;
    for (int j = tid; j < NTOK1; j += 256) {
        float v = s_cdf[j];
        mn = fminf(mn, v + ((v == 0.f) ? 1e8f : 0.f));
    }
    float ys_start = blk_red_min(mn, s_r8);
    for (int k = tid; k < NKEEP; k += 256) {
        float lin = (float)k / (float)(NKEEP - 1);
        s_ys[k] = ys_start + (lin * (float)(NKEEP - 1) - ys_start * (float)k) / (float)(NKEEP - 1);
    }
    __syncthreads();

    const int OFF = NTOK1 - NKEEP;  // 380 (diff = -380 in reference)
    for (int k = tid; k < NKEEP; k += 256) {
        float y = s_ys[k];
        int best = 0;
        float bd = fabsf(y - s_cdf[OFF]);
        for (int j = 1; j < NKEEP; ++j) {
            float d = fabsf(y - s_cdf[OFF + j]);
            if (d < bd) { bd = d; best = j; }
        }
        s_pick[k] = best + OFF;
    }
    __syncthreads();
    // sort picks, mark duplicates as NTOK1, sort again = get_unique_indices
    for (int i = tid; i < NKEEP; i += 256) {
        int pi = s_pick[i];
        int r = 0;
        for (int j = 0; j < NKEEP; ++j) {
            int pj = s_pick[j];
            r += (pj < pi) || (pj == pi && j < i);
        }
        s_u[r] = pi;
    }
    __syncthreads();
    for (int k = tid; k < NKEEP; k += 256) {
        int nxt = (k < NKEEP - 1) ? s_u[k + 1] : 1;
        s_u2[k] = (nxt == s_u[k]) ? NTOK1 : s_u[k];
    }
    __syncthreads();
    for (int i = tid; i < NKEEP; i += 256) {
        int pi = s_u2[i];
        int r = 0;
        for (int j = 0; j < NKEEP; ++j) {
            int pj = s_u2[j];
            r += (pj < pi) || (pj == pi && j < i);
        }
        s_u[r] = pi;  // final uniq
    }
    __syncthreads();

    if (tid == 0) {
        row_map[b * NSEL] = 0;
        pol_ws[b * NSEL] = 1.f;
        out_np[b * NSEL] = 1.f;
    }
    for (int t = tid; t < NKEEP; t += 256) {
        int u = s_u[t];
        int valid = (u < NTOK1);
        row_map[b * NSEL + 1 + t] = valid ? (s_idx[u] + 1) : -1;
        float p = valid ? 1.f : 0.f;
        pol_ws[b * NSEL + 1 + t] = p;
        out_np[b * NSEL + 1 + t] = p;
    }
}

// ------------------------------- row gather --------------------------------
__global__ void k_gather(const int* __restrict__ row_map,
                         const ushort_t* __restrict__ Ofull,
                         const float* __restrict__ x,
                         ushort_t* __restrict__ Osel,
                         float* __restrict__ xsel) {
    long i = (long)blockIdx.x * blockDim.x + threadIdx.x;
    long tot = (long)ROWS_SEL * C_;
    if (i >= tot) return;
    int s = (int)(i / C_), c = (int)(i % C_);
    int tok = row_map[s];
    int bb = s / NSEL;
    if (tok >= 0) {
        long src = ((long)bb * N_ + tok) * C_ + c;
        Osel[i] = Ofull[src];
        xsel[i] = x[src];
    } else {
        Osel[i] = 0;
        xsel[i] = 0.f;
    }
}

// ------------------------------ proj GEMM ----------------------------------
__global__ __launch_bounds__(256) void k_gemm_proj(const ushort_t* __restrict__ Osel,
                                                   const ushort_t* __restrict__ wT,
                                                   const float* __restrict__ bias,
                                                   const float* __restrict__ xsel,
                                                   const float* __restrict__ pol,
                                                   float* __restrict__ xb) {
    int wid = threadIdx.x >> 5, lane = threadIdx.x & 31, lm = lane & 15, half = lane >> 4;
    int wm = wid & 3, wn = wid >> 2;
    int arow0 = blockIdx.x * 128 + wm * 32;
    int bcol0 = blockIdx.y * 64 + wn * 32;
    Acc acc;
    gemm_core(Osel, C_, arow0, ROWS_SEL, wT, C_, bcol0, C_, C_, acc);
    for (int ti = 0; ti < 2; ++ti)
        for (int tj = 0; tj < 2; ++tj)
            for (int r = 0; r < 8; ++r) {
                int row = arow0 + ti * 16 + r + 8 * half;
                int col = bcol0 + tj * 16 + lm;
                if (row >= ROWS_SEL) continue;
                float p = pol[row];
                float out = (acc.c[ti][tj][r] + bias[col]) * p;
                xb[(long)row * C_ + col] = (xsel[(long)row * C_ + col] + out) * p;
            }
}

// ------------------------------ FC1 GEMM (GELU) ----------------------------
__global__ __launch_bounds__(256) void k_gemm_fc1(const ushort_t* __restrict__ hin,
                                                  const ushort_t* __restrict__ wT,
                                                  const float* __restrict__ bias,
                                                  ushort_t* __restrict__ aout) {
    int wid = threadIdx.x >> 5, lane = threadIdx.x & 31, lm = lane & 15, half = lane >> 4;
    int wm = wid & 3, wn = wid >> 2;
    int arow0 = blockIdx.x * 128 + wm * 32;
    int bcol0 = blockIdx.y * 64 + wn * 32;
    Acc acc;
    gemm_core(hin, C_, arow0, ROWS_SEL, wT, C_, bcol0, 4 * C_, C_, acc);
    for (int ti = 0; ti < 2; ++ti)
        for (int tj = 0; tj < 2; ++tj)
            for (int r = 0; r < 8; ++r) {
                int row = arow0 + ti * 16 + r + 8 * half;
                int col = bcol0 + tj * 16 + lm;
                if (row >= ROWS_SEL) continue;
                float v = acc.c[ti][tj][r] + bias[col];
                float g = 0.5f * v * (1.f + erff(v * 0.70710678118654752f));  // exact GELU
                aout[(long)row * (4 * C_) + col] = f2bf(g);
            }
}

// ------------------------------ FC2 GEMM -----------------------------------
__global__ __launch_bounds__(256) void k_gemm_fc2(const ushort_t* __restrict__ ain,
                                                  const ushort_t* __restrict__ wT,
                                                  const float* __restrict__ bias,
                                                  const float* __restrict__ xb,
                                                  const float* __restrict__ pol,
                                                  float* __restrict__ out) {
    int wid = threadIdx.x >> 5, lane = threadIdx.x & 31, lm = lane & 15, half = lane >> 4;
    int wm = wid & 3, wn = wid >> 2;
    int arow0 = blockIdx.x * 128 + wm * 32;
    int bcol0 = blockIdx.y * 64 + wn * 32;
    Acc acc;
    gemm_core(ain, 4 * C_, arow0, ROWS_SEL, wT, 4 * C_, bcol0, C_, 4 * C_, acc);
    for (int ti = 0; ti < 2; ++ti)
        for (int tj = 0; tj < 2; ++tj)
            for (int r = 0; r < 8; ++r) {
                int row = arow0 + ti * 16 + r + 8 * half;
                int col = bcol0 + tj * 16 + lm;
                if (row >= ROWS_SEL) continue;
                float p = pol[row];
                out[(long)row * C_ + col] =
                    (xb[(long)row * C_ + col] + acc.c[ti][tj][r] + bias[col]) * p;
            }
}

// ---------------------------------------------------------------------------
extern "C" void kernel_launch(void* const* d_in, const int* in_sizes, int n_in,
                              void* d_out, int out_size, void* d_ws, size_t ws_size,
                              hipStream_t stream) {
    const float* x      = (const float*)d_in[0];
    const float* policy = (const float*)d_in[1];
    const float* ln1_g  = (const float*)d_in[2];
    const float* ln1_b  = (const float*)d_in[3];
    const float* w_qkv  = (const float*)d_in[4];
    const float* w_proj = (const float*)d_in[5];
    const float* b_proj = (const float*)d_in[6];
    const float* ln2_g  = (const float*)d_in[7];
    const float* ln2_b  = (const float*)d_in[8];
    const float* w_fc1  = (const float*)d_in[9];
    const float* b_fc1  = (const float*)d_in[10];
    const float* w_fc2  = (const float*)d_in[11];
    const float* b_fc2  = (const float*)d_in[12];
    (void)in_sizes; (void)n_in; (void)out_size; (void)ws_size;

    float* out_xb = (float*)d_out;                       // [16,197,768]
    float* out_np = out_xb + (long)ROWS_SEL * C_;        // [16,197,1]

    char* w = (char*)d_ws;
    size_t off = 0;
    auto alloc = [&](size_t bytes) -> char* {
        char* p = w + off;
        off = (off + bytes + 255) & ~(size_t)255;
        return p;
    };
    ushort_t* wqkvT = (ushort_t*)alloc((size_t)3 * C_ * C_ * 2);          // [2304][768]
    ushort_t* wprojT = (ushort_t*)alloc((size_t)C_ * C_ * 2);             // [768][768]
    ushort_t* wfc1T = (ushort_t*)alloc((size_t)4 * C_ * C_ * 2);          // [3072][768]
    ushort_t* wfc2T = (ushort_t*)alloc((size_t)C_ * 4 * C_ * 2);          // [768][3072]
    ushort_t* xn    = (ushort_t*)alloc((size_t)ROWS_ALL * C_ * 2);
    ushort_t* qb    = (ushort_t*)alloc((size_t)B_ * H_ * N_ * D_ * 2);
    ushort_t* kb    = (ushort_t*)alloc((size_t)B_ * H_ * N_ * D_ * 2);
    ushort_t* vT    = (ushort_t*)alloc((size_t)B_ * H_ * D_ * NPAD * 2);
    ushort_t* attn  = (ushort_t*)alloc((size_t)B_ * H_ * N_ * NPAD * 2);
    float*    acls  = (float*)alloc((size_t)B_ * H_ * NPAD * 4);
    ushort_t* Ofull = (ushort_t*)alloc((size_t)B_ * N_ * C_ * 2);
    ushort_t* Osel  = (ushort_t*)alloc((size_t)ROWS_SEL * C_ * 2);
    float*    xsel  = (float*)alloc((size_t)ROWS_SEL * C_ * 4);
    float*    xbuf  = (float*)alloc((size_t)ROWS_SEL * C_ * 4);
    ushort_t* hbuf  = (ushort_t*)alloc((size_t)ROWS_SEL * C_ * 2);
    ushort_t* abuf  = (ushort_t*)alloc((size_t)ROWS_SEL * 4 * C_ * 2);
    int*      rmap  = (int*)alloc((size_t)ROWS_SEL * 4);
    float*    polw  = (float*)alloc((size_t)ROWS_SEL * 4);

    // 1. weight cast + transpose to bf16 [N][K]
    {
        long t;
        t = (long)C_ * 3 * C_;
        k_cvt_T<<<dim3((unsigned)((t + 255) / 256)), dim3(256), 0, stream>>>(w_qkv, wqkvT, C_, 3 * C_);
        t = (long)C_ * C_;
        k_cvt_T<<<dim3((unsigned)((t + 255) / 256)), dim3(256), 0, stream>>>(w_proj, wprojT, C_, C_);
        t = (long)C_ * 4 * C_;
        k_cvt_T<<<dim3((unsigned)((t + 255) / 256)), dim3(256), 0, stream>>>(w_fc1, wfc1T, C_, 4 * C_);
        t = (long)4 * C_ * C_;
        k_cvt_T<<<dim3((unsigned)((t + 255) / 256)), dim3(256), 0, stream>>>(w_fc2, wfc2T, 4 * C_, C_);
    }
    // 2. zero vT (token padding must be 0 for K-padded attn@v)
    k_zero_us<<<dim3(1024), dim3(256), 0, stream>>>(vT, (long)B_ * H_ * D_ * NPAD);
    // 3. LN1 -> bf16
    k_layernorm<<<dim3(ROWS_ALL), dim3(256), 0, stream>>>(x, ln1_g, ln1_b, xn, C_);
    // 4. QKV GEMM  M=9232 N=2304 K=768
    k_gemm_qkv<<<dim3((ROWS_ALL + 127) / 128, (3 * C_) / 64), dim3(256), 0, stream>>>(
        xn, wqkvT, policy, qb, kb, vT);
    // 5. scores + softmax-with-policy
    k_scores_softmax<<<dim3((N_ + 15) / 16, H_, B_), dim3(256), 0, stream>>>(
        qb, kb, policy, attn, acls);
    // 6. attn @ v   per (b,h): M=577 N=64 K=608
    k_attn_v<<<dim3((N_ + 127) / 128, H_, B_), dim3(256), 0, stream>>>(attn, vT, Ofull);
    // 7. token selection (sig/sort/cdf/pick/unique) -> row map + new_policy
    k_select<<<dim3(B_), dim3(256), 0, stream>>>(acls, vT, rmap, polw, out_np);
    // 8. gather selected rows
    {
        long t = (long)ROWS_SEL * C_;
        k_gather<<<dim3((unsigned)((t + 255) / 256)), dim3(256), 0, stream>>>(
            rmap, Ofull, x, Osel, xsel);
    }
    // 9. proj GEMM + residual + policy  M=3152 N=768 K=768
    k_gemm_proj<<<dim3((ROWS_SEL + 127) / 128, C_ / 64), dim3(256), 0, stream>>>(
        Osel, wprojT, b_proj, xsel, polw, xbuf);
    // 10. LN2 -> bf16
    k_layernorm<<<dim3(ROWS_SEL), dim3(256), 0, stream>>>(xbuf, ln2_g, ln2_b, hbuf, C_);
    // 11. FC1 GEMM + GELU  M=3152 N=3072 K=768
    k_gemm_fc1<<<dim3((ROWS_SEL + 127) / 128, (4 * C_) / 64), dim3(256), 0, stream>>>(
        hbuf, wfc1T, b_fc1, abuf);
    // 12. FC2 GEMM + residual + policy -> d_out  M=3152 N=768 K=3072
    k_gemm_fc2<<<dim3((ROWS_SEL + 127) / 128, C_ / 64), dim3(256), 0, stream>>>(
        abuf, wfc2T, b_fc2, xbuf, polw, out_xb);
}
